// LSTM_35029753266825
// MI455X (gfx1250) — matrix operand order repaired and checked by reading, not compiled, not run on hardware
//
#include <hip/hip_runtime.h>
#include <math.h>

// LSTM: B=64, T=512, I=1024, H=1024.
// Phase 1: xproj = x @ [wxi|wxf|wxg|wxo] + (bx+bh) -> stored [T][B][4H] fp32 (ws)
// Phase 2: ONE persistent kernel, 64 WGs; each WG stages its 128KB whT slice
//          into LDS (B-fragment layout) once, then runs all 512 steps with a
//          device-wide atomic barrier between steps. c-state stays in registers.
#define BB 64
#define TT 512
#define II 1024
#define HH 1024
#define FH 4096    // 4*H
#define BTT 32768  // B*T
#define NWG 64     // persistent workgroups (16 columns of H each)

typedef __attribute__((ext_vector_type(16))) __bf16 v16bf;
typedef __attribute__((ext_vector_type(8)))  float  v8f;
typedef __attribute__((ext_vector_type(8)))  unsigned short v8u16;

__device__ __forceinline__ unsigned short f2bf(float x) {
  unsigned int u = __float_as_uint(x);
  u += 0x7FFFu + ((u >> 16) & 1u);   // round-to-nearest-even
  return (unsigned short)(u >> 16);
}

// 16-bit A fragment (16x32, MxK): lane L = row M=L%16; element e holds
// K = k0 + (L/16)*16 + e (16 contiguous).
__device__ __forceinline__ v16bf load_a_frag(const unsigned short* p) {
  union { v16bf v; v8u16 h[2]; } f;
  f.h[0] = *reinterpret_cast<const v8u16*>(p);
  f.h[1] = *reinterpret_cast<const v8u16*>(p + 8);
  return f.v;
}
// 16-bit B fragment (32x16, KxN) from pre-transposed weights (row=n, col=k):
// lane L = col N=L%16; element e holds K = k0 + (L/16)*16 + e (16 contiguous).
__device__ __forceinline__ v16bf load_b_frag(const unsigned short* p) {
  union { v16bf v; v8u16 h[2]; } f;
  f.h[0] = *reinterpret_cast<const v8u16*>(p);
  f.h[1] = *reinterpret_cast<const v8u16*>(p + 8);
  return f.v;
}

__device__ __forceinline__ float sig_(float x) { return 1.0f / (1.0f + __expf(-x)); }

// ---------------- prep kernels ----------------
__global__ __launch_bounds__(256) void cvt_bf16_kernel(
    const float* __restrict__ in, unsigned short* __restrict__ out, int n) {
  int i = blockIdx.x * 256 + threadIdx.x;
  if (i < n) out[i] = f2bf(in[i]);
}

// out[n*1024 + k] = bf16(w[k*1024 + n])   (1024x1024 transpose+convert)
__global__ __launch_bounds__(256) void transpose_cvt_kernel(
    const float* __restrict__ w, unsigned short* __restrict__ out) {
  int i = blockIdx.x * 256 + threadIdx.x;     // 1M threads
  int n = i >> 10;
  int k = i & 1023;
  out[(size_t)n * 1024 + k] = f2bf(w[(size_t)k * 1024 + n]);
}

__global__ __launch_bounds__(256) void bias_kernel(
    const float* bxi, const float* bhi, const float* bxf, const float* bhf,
    const float* bxg, const float* bhg, const float* bxo, const float* bho,
    float* __restrict__ bias) {
  int i = blockIdx.x * 256 + threadIdx.x;     // 4096
  int g = i >> 10, j = i & 1023;
  const float* bx = (g == 0) ? bxi : (g == 1) ? bxf : (g == 2) ? bxg : bxo;
  const float* bh = (g == 0) ? bhi : (g == 1) ? bhf : (g == 2) ? bhg : bho;
  bias[i] = bx[j] + bh[j];
}

__global__ __launch_bounds__(256) void init_state_kernel(
    unsigned short* __restrict__ h0, int* __restrict__ bar) {
  int i = blockIdx.x * 256 + threadIdx.x;     // 65536
  h0[i] = 0;
  if (i == 0) *bar = 0;
}

// ---------------- phase 1: xproj GEMM (32768x1024 @ 1024x4096) ----------------
// 8 waves/WG; each wave-job = 64(M) x 32(N); output permuted to [T][B][4H].
__global__ __launch_bounds__(256) void xproj_gemm_kernel(
    const unsigned short* __restrict__ xbf,   // BTT x II bf16 (rows = b*T + t)
    const unsigned short* __restrict__ wxT,   // FH  x II bf16 (transposed)
    const float* __restrict__ bias,           // FH
    float* __restrict__ xproj) {              // [T][B][FH] fp32
  const int lane = threadIdx.x & 31;
  const int wave = threadIdx.x >> 5;
  const int ln = lane & 15;
  const int lh = lane >> 4;
  const int job  = blockIdx.x * 8 + wave;     // 65536 wave-jobs
  const int mJob = job >> 7;                  // BTT/64 = 512
  const int nJob = job & 127;                 // FH/32  = 128
  const int m_base = mJob * 64;
  const int n_base = nJob * 32;

  v8f zero8;
#pragma unroll
  for (int r = 0; r < 8; ++r) zero8[r] = 0.0f;
  v8f acc[4][2];
#pragma unroll
  for (int mt = 0; mt < 4; ++mt)
#pragma unroll
    for (int j = 0; j < 2; ++j) acc[mt][j] = zero8;

  const unsigned short* arow[4];
#pragma unroll
  for (int mt = 0; mt < 4; ++mt)
    arow[mt] = xbf + (size_t)(m_base + mt * 16 + ln) * II;
  const unsigned short* brow[2];
#pragma unroll
  for (int j = 0; j < 2; ++j)
    brow[j] = wxT + (size_t)(n_base + j * 16 + ln) * II;

  for (int k0 = 0; k0 < II; k0 += 32) {
    v16bf bf[2];
#pragma unroll
    for (int j = 0; j < 2; ++j) bf[j] = load_b_frag(brow[j] + k0 + lh * 16);
#pragma unroll
    for (int mt = 0; mt < 4; ++mt) {
      v16bf af = load_a_frag(arow[mt] + k0 + lh * 16);
#pragma unroll
      for (int j = 0; j < 2; ++j)
        acc[mt][j] = __builtin_amdgcn_wmma_f32_16x16x32_bf16(
            false, af, false, bf[j], (short)0, acc[mt][j], false, false);
    }
  }
  // C/D layout: element r of lane L -> (M = r + 8*(L/16), N = L%16)
#pragma unroll
  for (int mt = 0; mt < 4; ++mt) {
#pragma unroll
    for (int j = 0; j < 2; ++j) {
      const int n = n_base + j * 16 + ln;
      const float bn = bias[n];
#pragma unroll
      for (int r = 0; r < 8; ++r) {
        const int m = m_base + mt * 16 + r + lh * 8;   // m = b*T + t
        const size_t orow = (size_t)(m & 511) * BB + (m >> 9);  // -> t*B + b
        xproj[orow * FH + n] = acc[mt][j][r] + bn;
      }
    }
  }
}

// ---------------- phase 2: persistent recurrent kernel ----------------
// Grid = 64 WGs x 128 threads (4 waves = 4 M-tiles of batch 64).
// WG b owns columns [b*16, b*16+16) of H across ALL 4 gates; its whT slice
// (4 x 16 x 1024 bf16 = 128 KB) is staged into LDS in B-fragment layout:
//   chunk c = ((g*32 + ki)*2 + half)*32 + lane, 8 u16 each (lane stride 16B).
__global__ __launch_bounds__(128) void lstm_persistent_kernel(
    const unsigned short* __restrict__ whT,    // FH x HH bf16 (transposed)
    const float* __restrict__ xproj,           // [T][B][FH] fp32
    unsigned short* __restrict__ h0buf,        // BB x HH bf16 (zeroed)
    unsigned short* __restrict__ h1buf,        // BB x HH bf16
    float* __restrict__ hout,                  // d_out: final h fp32
    float* __restrict__ cout,                  // d_out+65536: final c fp32
    int* __restrict__ bar) {                   // global barrier counter (=0)
  __shared__ unsigned short lds_wh[65536];     // 128 KB

  const int lane = threadIdx.x & 31;
  const int wave = threadIdx.x >> 5;           // 0..3 = M-tile
  const int ln = lane & 15;
  const int lh = lane >> 4;
  const int m0 = wave * 16;
  const int nb = blockIdx.x * 16;

  // ---- one-time LDS staging of this WG's weight slice ----
  for (int c = threadIdx.x; c < 8192; c += 128) {
    const int lane_s = c & 31;
    const int half   = (c >> 5) & 1;
    const int ki     = (c >> 6) & 31;
    const int g      = c >> 11;
    const int n      = nb + (lane_s & 15);
    const int k      = ki * 32 + (lane_s >> 4) * 16 + half * 8;
    *reinterpret_cast<v8u16*>(&lds_wh[(size_t)c * 8]) =
        *reinterpret_cast<const v8u16*>(&whT[(size_t)(g * HH + n) * HH + k]);
  }
  __syncthreads();

  float creg[8];                               // c-state lives in registers
#pragma unroll
  for (int r = 0; r < 8; ++r) creg[r] = 0.0f;

  v8f zero8;
#pragma unroll
  for (int r = 0; r < 8; ++r) zero8[r] = 0.0f;

  const int n = nb + ln;
  unsigned short* hb[2] = {h0buf, h1buf};

  for (int t = 0; t < TT; ++t) {
    const unsigned short* hprev = hb[t & 1];
    unsigned short* hnext = hb[(t + 1) & 1];

    v8f acc[4];
#pragma unroll
    for (int g = 0; g < 4; ++g) acc[g] = zero8;

    const unsigned short* arow = hprev + (size_t)(m0 + ln) * HH;
    for (int ki = 0; ki < 32; ++ki) {
      v16bf af = load_a_frag(arow + ki * 32 + lh * 16);
#pragma unroll
      for (int g = 0; g < 4; ++g) {
        union { v16bf v; v8u16 h[2]; } bfr;
        const unsigned short* p = &lds_wh[((size_t)(g * 32 + ki) * 64 + lane) * 8];
        bfr.h[0] = *reinterpret_cast<const v8u16*>(p);
        bfr.h[1] = *reinterpret_cast<const v8u16*>(p + 256);
        acc[g] = __builtin_amdgcn_wmma_f32_16x16x32_bf16(
            false, af, false, bfr.v, (short)0, acc[g], false, false);
      }
    }

    const float* xp = xproj + (size_t)t * BB * FH;   // contiguous 1MB block
#pragma unroll
    for (int r = 0; r < 8; ++r) {
      const int m = m0 + r + lh * 8;                 // batch index
      const float* xpm = xp + (size_t)m * FH;
      const float gi = sig_(acc[0][r] + xpm[0 * HH + n]);
      const float gf = sig_(acc[1][r] + xpm[1 * HH + n]);
      const float gg = tanhf(acc[2][r] + xpm[2 * HH + n]);
      const float go = sig_(acc[3][r] + xpm[3 * HH + n]);
      creg[r] = gf * creg[r] + gi * gg;
      const float hn = go * tanhf(creg[r]);
      hnext[(size_t)m * HH + n] = f2bf(hn);
      if (t == TT - 1) {                             // uniform branch
        hout[(size_t)m * HH + n] = hn;
        cout[(size_t)m * HH + n] = creg[r];
      }
    }

    // ---- device-wide barrier (monotonic counter, no sense reversal) ----
    __threadfence();
    __syncthreads();
    if (threadIdx.x == 0) {
      atomicAdd(bar, 1);
      const int target = NWG * (t + 1);
      while (__hip_atomic_load(bar, __ATOMIC_RELAXED, __HIP_MEMORY_SCOPE_AGENT) <
             target)
        __builtin_amdgcn_s_sleep(2);
    }
    __syncthreads();
    __threadfence();
  }
}

// ---------------- launch ----------------
extern "C" void kernel_launch(void* const* d_in, const int* in_sizes, int n_in,
                              void* d_out, int out_size, void* d_ws, size_t ws_size,
                              hipStream_t stream) {
  (void)in_sizes; (void)n_in; (void)out_size; (void)ws_size;
  // input order: x, wxi, whi, wxf, whf, wxg, whg, wxo, who,
  //              bxi, bhi, bxf, bhf, bxg, bhg, bxo, bho
  const float* x = (const float*)d_in[0];
  const float* wx[4] = {(const float*)d_in[1], (const float*)d_in[3],
                        (const float*)d_in[5], (const float*)d_in[7]};
  const float* wh[4] = {(const float*)d_in[2], (const float*)d_in[4],
                        (const float*)d_in[6], (const float*)d_in[8]};

  // workspace layout (needs ~593 MiB)
  char* base = (char*)d_ws;
  constexpr size_t XPROJ_OFF = 0;                                    // 512 MiB fp32
  constexpr size_t XBF_OFF   = XPROJ_OFF + (size_t)BTT * FH * 4;     // 64 MiB bf16
  constexpr size_t WXT_OFF   = XBF_OFF + (size_t)BTT * II * 2;       // 8 MiB
  constexpr size_t WHT_OFF   = WXT_OFF + (size_t)FH * II * 2;        // 8 MiB
  constexpr size_t BIAS_OFF  = WHT_OFF + (size_t)FH * HH * 2;        // 16 KiB
  constexpr size_t H0_OFF    = BIAS_OFF + (size_t)FH * 4;            // 128 KiB
  constexpr size_t H1_OFF    = H0_OFF + (size_t)BB * HH * 2;         // 128 KiB
  constexpr size_t BAR_OFF   = H1_OFF + (size_t)BB * HH * 2;         // 4 B

  float* xproj          = (float*)(base + XPROJ_OFF);
  unsigned short* xbf   = (unsigned short*)(base + XBF_OFF);
  unsigned short* wxT   = (unsigned short*)(base + WXT_OFF);
  unsigned short* whT   = (unsigned short*)(base + WHT_OFF);
  float* bias           = (float*)(base + BIAS_OFF);
  unsigned short* hb0   = (unsigned short*)(base + H0_OFF);
  unsigned short* hb1   = (unsigned short*)(base + H1_OFF);
  int* bar              = (int*)(base + BAR_OFF);

  float* hout = (float*)d_out;            // final h (64x1024)
  float* cout = hout + (size_t)BB * HH;   // final c (64x1024)

  // prep: convert x, transpose+convert weights, fold biases, zero h0/barrier
  cvt_bf16_kernel<<<(BTT * II) / 256, 256, 0, stream>>>(x, xbf, BTT * II);
  for (int g = 0; g < 4; ++g) {
    transpose_cvt_kernel<<<(II * HH) / 256, 256, 0, stream>>>(
        wx[g], wxT + (size_t)g * HH * II);
    transpose_cvt_kernel<<<(HH * HH) / 256, 256, 0, stream>>>(
        wh[g], whT + (size_t)g * HH * HH);
  }
  bias_kernel<<<FH / 256, 256, 0, stream>>>(
      (const float*)d_in[9],  (const float*)d_in[10],
      (const float*)d_in[11], (const float*)d_in[12],
      (const float*)d_in[13], (const float*)d_in[14],
      (const float*)d_in[15], (const float*)d_in[16], bias);
  init_state_kernel<<<(BB * HH) / 256, 256, 0, stream>>>(hb0, bar);

  // phase 1: big input-projection GEMM
  xproj_gemm_kernel<<<8192, 256, 0, stream>>>(xbf, wxT, bias, xproj);

  // phase 2: one persistent kernel runs all 512 steps (weights LDS-resident)
  lstm_persistent_kernel<<<NWG, 128, 0, stream>>>(
      whT, xproj, hb0, hb1, hout, cout, bar);
}
